// TransformerEncoderLayer_16681652978110
// MI455X (gfx1250) — matrix operand order repaired and checked by reading, not compiled, not run on hardware
//
#include <hip/hip_runtime.h>

// ---------------- problem constants ----------------
#define TSEQ    1024
#define BBATCH  4
#define NHEAD   8
#define DQKV    96
#define DMODEL  768
#define DFF     3072
#define MAXREL  100
#define NUMEMB  199            // 2*MAXREL-1
#define MROWS   (TSEQ*BBATCH)  // 4096 flattened (t,b) rows
#define WCOLS   256            // banded-attention window columns
#define RCOLS   208            // 13 tiles of relative-offset columns (covers 0..198)

typedef __attribute__((ext_vector_type(16))) __bf16 v16bf;
typedef __attribute__((ext_vector_type(8)))  __bf16 bf16x8;
typedef __attribute__((ext_vector_type(4)))  __bf16 bf16x4;
typedef __attribute__((ext_vector_type(8)))  float  f32x8;
typedef __attribute__((ext_vector_type(4)))  float  f32x4;

static __device__ __forceinline__ f32x8 wmma_bf16(v16bf a, v16bf b, f32x8 c) {
    return __builtin_amdgcn_wmma_f32_16x16x32_bf16(
        false, a, false, b, (short)0, c, false, false);
}

// A-operand (16-bit A 16x32): lane half h:
//   elements 0..15 -> K = h*16 + e       (16 contiguous)
// B-operand: K = h*16 + e                (16 contiguous)
static __device__ __forceinline__ int b_koff(int half, int e) {
    return half * 16 + e;
}

// ---- bf16-source fragment loads: raw b128s, no conversion ----
static __device__ __forceinline__ v16bf load_a_bf(const __bf16* row, int half) {
    bf16x8 lo = *(const bf16x8*)(row + half * 16);
    bf16x8 hi = *(const bf16x8*)(row + half * 16 + 8);
    v16bf f;
#pragma unroll
    for (int e = 0; e < 8; ++e) { f[e] = lo[e]; f[e + 8] = hi[e]; }
    return f;
}
static __device__ __forceinline__ v16bf load_b_bf(const __bf16* row, int half) {
    bf16x8 lo = *(const bf16x8*)(row + half * 16);
    bf16x8 hi = *(const bf16x8*)(row + half * 16 + 8);
    v16bf f;
#pragma unroll
    for (int e = 0; e < 8; ++e) { f[e] = lo[e]; f[e + 8] = hi[e]; }
    return f;
}
// ---- fp32-source fragment loads (rel_emb and LDS probs) ----
static __device__ __forceinline__ void put4(v16bf& f, int e0, f32x4 v) {
    f[e0 + 0] = (__bf16)v[0]; f[e0 + 1] = (__bf16)v[1];
    f[e0 + 2] = (__bf16)v[2]; f[e0 + 3] = (__bf16)v[3];
}
static __device__ __forceinline__ v16bf load_a_f32(const float* row, int half) {
    const f32x4* p0 = (const f32x4*)(row + half * 16);
    const f32x4* p1 = (const f32x4*)(row + half * 16 + 8);
    v16bf f;
    put4(f, 0, p0[0]); put4(f, 4, p0[1]);
    put4(f, 8, p1[0]); put4(f, 12, p1[1]);
    return f;
}
static __device__ __forceinline__ v16bf load_b_f32(const float* row, int half) {
    const f32x4* p = (const f32x4*)(row + half * 16);
    v16bf f;
    put4(f, 0, p[0]); put4(f, 4, p[1]); put4(f, 8, p[2]); put4(f, 12, p[3]);
    return f;
}

// =====================================================================
// Weight / activation prep kernels (one-time conversions per launch).
// =====================================================================
__global__ __launch_bounds__(256)
void cvt_bf16_kernel(const float* __restrict__ in, __bf16* __restrict__ out, int n4)
{
    const int i = blockIdx.x * 256 + threadIdx.x;
    if (i >= n4) return;
    f32x4 v = ((const f32x4*)in)[i];
    bf16x4 o;
    o[0] = (__bf16)v[0]; o[1] = (__bf16)v[1]; o[2] = (__bf16)v[2]; o[3] = (__bf16)v[3];
    ((bf16x4*)out)[i] = o;
}
// w (H,768,96) -> Bt[(h*96+a)*768 + f], bf16
__global__ __launch_bounds__(256)
void whead_kernel(const float* __restrict__ w, __bf16* __restrict__ out)
{
    const int i = blockIdx.x * 256 + threadIdx.x;   // 589824 total
    const int n = i / DMODEL, k = i - n * DMODEL;
    const int h = n / DQKV,  a = n - h * DQKV;
    out[i] = (__bf16)w[(size_t)h * (DMODEL * DQKV) + (size_t)k * DQKV + a];
}
// w_o (768,768) k-major -> Bt[n*768 + k], bf16 (transpose)
__global__ __launch_bounds__(256)
void wo_t_kernel(const float* __restrict__ w, __bf16* __restrict__ out)
{
    const int i = blockIdx.x * 256 + threadIdx.x;
    const int n = i / DMODEL, k = i - n * DMODEL;
    out[i] = (__bf16)w[(size_t)k * DMODEL + n];
}

// =====================================================================
// Register-blocked bf16 WMMA GEMM: one wave owns a 64x64 block (4x4 tiles).
// Per 32-K chunk: 8 + 8 b128 loads -> 16 WMMAs, no conversions.
//   A:  bf16 row-major MxK (lda, z-stride zA)
//   Bt: bf16, Bt[n*K + k]  (uniform pre-transposed weight layout)
//   Out[m,n] = relu?(acc + bias[n]) + Res[m,n]; OUTBF selects bf16/f32 store.
//   M%64==0, N%64==0, K%32==0.
// =====================================================================
template<bool RELU, bool HASBIAS, bool HASRES, bool OUTBF>
__global__ __launch_bounds__(256)
void gemm_wmma_kernel(const __bf16* __restrict__ A, int lda, long long zA,
                      const __bf16* __restrict__ Bt,
                      const float* __restrict__ bias,
                      const float* __restrict__ Res,
                      void* __restrict__ OutV, int ldo, long long zO,
                      int M, int N, int K)
{
    const int lane = threadIdx.x & 31;
    const int wave = threadIdx.x >> 5;
    const int half = lane >> 4;
    const int lm   = lane & 15;

    const int tilesN = N >> 6;
    const int tile   = blockIdx.x * 8 + wave;
    if (tile >= (M >> 6) * tilesN) return;          // wave-uniform
    const int mt = tile / tilesN;
    const int nt = tile - mt * tilesN;
    const int m0 = mt << 6;
    const int n0 = nt << 6;

    const __bf16* Ab = A + (size_t)blockIdx.z * (size_t)zA;

    const __bf16* Arow[4];
    const __bf16* Brow[4];
#pragma unroll
    for (int i = 0; i < 4; ++i) {
        Arow[i] = Ab + (size_t)(m0 + i * 16 + lm) * lda;
        Brow[i] = Bt + (size_t)(n0 + i * 16 + lm) * K;
    }

    f32x8 acc[4][4] = {};
    for (int k0 = 0; k0 < K; k0 += 32) {
        v16bf af[4], bf[4];
#pragma unroll
        for (int i = 0; i < 4; ++i) af[i] = load_a_bf(Arow[i] + k0, half);
#pragma unroll
        for (int j = 0; j < 4; ++j) bf[j] = load_b_bf(Brow[j] + k0, half);
#pragma unroll
        for (int i = 0; i < 4; ++i)
#pragma unroll
            for (int j = 0; j < 4; ++j)
                acc[i][j] = wmma_bf16(af[i], bf[j], acc[i][j]);
    }

    float bv[4];
#pragma unroll
    for (int j = 0; j < 4; ++j) bv[j] = HASBIAS ? bias[n0 + j * 16 + lm] : 0.f;

#pragma unroll
    for (int i = 0; i < 4; ++i) {
#pragma unroll
        for (int j = 0; j < 4; ++j) {
            const int n = n0 + j * 16 + lm;
#pragma unroll
            for (int r = 0; r < 8; ++r) {
                const int row = m0 + i * 16 + r + 8 * half;   // C: M = r + 8*half
                float v = acc[i][j][r] + bv[j];
                if (RELU)   v = fmaxf(v, 0.f);
                if (HASRES) v += Res[(size_t)row * ldo + n];
                const size_t oidx = (size_t)blockIdx.z * (size_t)zO + (size_t)row * ldo + n;
                if (OUTBF) ((__bf16*)OutV)[oidx] = (__bf16)v;
                else       ((float*)OutV)[oidx]  = v;
            }
        }
    }
}

// =====================================================================
// Banded relative-position attention (exact: -1e8 rel-pos masking makes
// softmax a |k-q|<=99 band since T > MAX_REL).
// One wave per (b, h, 16-row q-tile); 256-column aligned window.
// Q/K/V in bf16 (b, t, 768); output bf16 in (t*B+b, 768).
// =====================================================================
__global__ __launch_bounds__(32)
void attn_kernel(const __bf16* __restrict__ Q, const __bf16* __restrict__ Km,
                 const __bf16* __restrict__ V, const float* __restrict__ emb,
                 __bf16* __restrict__ Oout)
{
    __shared__ float Rld[16 * RCOLS];   // rel-pos logits per (q-row, rel offset)
    __shared__ float Lg[16 * WCOLS];    // logits -> probs

    const int lane = threadIdx.x;
    const int half = lane >> 4;
    const int lm   = lane & 15;
    const int q0   = blockIdx.x * 16;
    const int h    = blockIdx.y;
    const int b    = blockIdx.z;
    const size_t base = ((size_t)b * TSEQ) * DMODEL + (size_t)h * DQKV;

    // ---- 1) Q fragments (A-operand): raw bf16 b128 loads ----
    v16bf qf[3];
    {
        const __bf16* qrow = Q + base + (size_t)(q0 + lm) * DMODEL;
#pragma unroll
        for (int c = 0; c < 3; ++c) qf[c] = load_a_bf(qrow + c * 32, half);
    }

    // ---- 2) R = Qtile x rel_emb^T (13 n-tiles x 3 k-chunks) ----
    const float* eh = emb + (size_t)h * NUMEMB * DQKV;
    for (int mt = 0; mt < 13; ++mt) {
        const int mcol  = mt * 16 + lm;
        const bool mok  = (mcol < NUMEMB);
        const float* er = eh + (size_t)mcol * DQKV;
        f32x8 acc = {};
#pragma unroll
        for (int c = 0; c < 3; ++c) {
            v16bf bf;
            if (mok) bf = load_b_f32(er + c * 32, half);
            else {
#pragma unroll
                for (int e = 0; e < 16; ++e) bf[e] = (__bf16)0.f;
            }
            acc = wmma_bf16(qf[c], bf, acc);
        }
#pragma unroll
        for (int r = 0; r < 8; ++r)
            Rld[(r + 8 * half) * RCOLS + mt * 16 + lm] = acc[r];
    }

    // ---- 3) logits over the 256-column window (Q x K^T) ----
    const int   kstart = q0 - 112;
    const float scale  = 0.1020620726f;              // 1/sqrt(96)
    for (int j = 0; j < 16; ++j) {
        const int  kcol = kstart + j * 16 + lm;      // key index owned by lane
        const bool kok  = (kcol >= 0) && (kcol < TSEQ);
        const __bf16* kr = Km + base + (size_t)(kok ? kcol : 0) * DMODEL;
        f32x8 acc = {};
#pragma unroll
        for (int c = 0; c < 3; ++c) {
            v16bf bf;
            if (kok) bf = load_b_bf(kr + c * 32, half);
            else {
#pragma unroll
                for (int e = 0; e < 16; ++e) bf[e] = (__bf16)0.f;
            }
            acc = wmma_bf16(qf[c], bf, acc);
        }
#pragma unroll
        for (int r = 0; r < 8; ++r) {
            const int row = r + 8 * half;
            const int rel = kcol - (q0 + row) + (MAXREL - 1);
            float v;
            if (!kok || rel < 0 || rel >= NUMEMB) v = -1e8f;   // exact mask
            else v = acc[r] * scale + Rld[row * RCOLS + rel];
            Lg[row * WCOLS + j * 16 + lm] = v;
        }
    }

    // ---- 4) row softmax (16 rows, one lane each; wave re-converges) ----
    if (lane < 16) {
        float* rowp = &Lg[lane * WCOLS];
        float mx = -3.4e38f;
        for (int i = 0; i < WCOLS; ++i) mx = fmaxf(mx, rowp[i]);
        float sum = 0.f;
        for (int i = 0; i < WCOLS; ++i) { float p = __expf(rowp[i] - mx); rowp[i] = p; sum += p; }
        const float inv = 1.f / sum;
        for (int i = 0; i < WCOLS; ++i) rowp[i] *= inv;
    }

    // ---- 5) O = P x V (8 K-chunks x 6 N-tiles); P via ds_load_b128 ----
    for (int nt = 0; nt < 6; ++nt) {
        f32x8 acc = {};
        for (int cc = 0; cc < 8; ++cc) {
            const float* prow = &Lg[lm * WCOLS + cc * 32];
            v16bf pa = load_a_f32(prow, half);
            v16bf vb;
#pragma unroll
            for (int e = 0; e < 16; ++e) {
                const int kcol = kstart + cc * 32 + b_koff(half, e);
                vb[e] = (kcol >= 0 && kcol < TSEQ)
                            ? V[base + (size_t)kcol * DMODEL + nt * 16 + lm]
                            : (__bf16)0.f;
            }
            acc = wmma_bf16(pa, vb, acc);
        }
#pragma unroll
        for (int r = 0; r < 8; ++r) {
            const int row = r + 8 * half;
            Oout[((size_t)(q0 + row) * BBATCH + b) * DMODEL + (size_t)h * DQKV + nt * 16 + lm]
                = (__bf16)acc[r];
        }
    }
}

// =====================================================================
// LayerNorm over last dim (768), one block per row.
// DUAL: also emit a bf16 copy (feeds the next GEMM's A operand).
// =====================================================================
template<bool DUAL>
__global__ __launch_bounds__(256)
void ln_kernel(const float* __restrict__ in, const float* __restrict__ g,
               const float* __restrict__ bta, float* __restrict__ out,
               __bf16* __restrict__ out_bf)
{
    __shared__ float red[512];
    const int row = blockIdx.x;
    const float* x = in + (size_t)row * DMODEL;
    float s = 0.f, s2 = 0.f;
    for (int i = threadIdx.x; i < DMODEL; i += 256) { float v = x[i]; s += v; s2 += v * v; }
    red[threadIdx.x] = s; red[256 + threadIdx.x] = s2;
    __syncthreads();
    for (int off = 128; off > 0; off >>= 1) {
        if (threadIdx.x < off) {
            red[threadIdx.x]       += red[threadIdx.x + off];
            red[256 + threadIdx.x] += red[256 + threadIdx.x + off];
        }
        __syncthreads();
    }
    const float mean = red[0] * (1.f / DMODEL);
    const float var  = red[256] * (1.f / DMODEL) - mean * mean;
    const float inv  = rsqrtf(var + 1e-5f);
    for (int i = threadIdx.x; i < DMODEL; i += 256) {
        const float v = g[i] * ((x[i] - mean) * inv) + bta[i];
        out[(size_t)row * DMODEL + i] = v;
        if (DUAL) out_bf[(size_t)row * DMODEL + i] = (__bf16)v;
    }
}

// =====================================================================
// Host-side orchestration
// =====================================================================
extern "C" void kernel_launch(void* const* d_in, const int* in_sizes, int n_in,
                              void* d_out, int out_size, void* d_ws, size_t ws_size,
                              hipStream_t stream)
{
    (void)in_sizes; (void)n_in; (void)out_size; (void)ws_size;
    const float* src     = (const float*)d_in[0];
    const float* w_q     = (const float*)d_in[1];
    const float* w_k     = (const float*)d_in[2];
    const float* w_v     = (const float*)d_in[3];
    const float* w_o     = (const float*)d_in[4];
    const float* rel_emb = (const float*)d_in[5];
    const float* lin1_w  = (const float*)d_in[6];   // (3072,768)  == Bt layout
    const float* lin1_b  = (const float*)d_in[7];
    const float* lin2_w  = (const float*)d_in[8];   // (768,3072)  == Bt layout
    const float* lin2_b  = (const float*)d_in[9];
    const float* ln1_g   = (const float*)d_in[10];
    const float* ln1_b   = (const float*)d_in[11];
    const float* ln2_g   = (const float*)d_in[12];
    const float* ln2_b   = (const float*)d_in[13];

    // ---- workspace layout ----
    const size_t SEG  = (size_t)BBATCH * TSEQ * DMODEL;   // 3,145,728 elements
    const size_t WSM  = (size_t)DMODEL * DMODEL;          // 589,824
    const size_t WFF  = (size_t)DFF * DMODEL;             // 2,359,296
    char* p = (char*)d_ws;
    auto take = [&](size_t bytes) { char* q = p; p += (bytes + 255) & ~(size_t)255; return q; };
    __bf16* src_bf  = (__bf16*)take(SEG * 2);
    __bf16* Qws     = (__bf16*)take(SEG * 2);     // (b, t, h*96+a)
    __bf16* Kws     = (__bf16*)take(SEG * 2);
    __bf16* Vws     = (__bf16*)take(SEG * 2);
    __bf16* attn    = (__bf16*)take(SEG * 2);     // (t*B+b, 768)
    __bf16* xbf     = (__bf16*)take(SEG * 2);     // bf16 copy of LN1 output
    float*  xpre    = (float*) take(SEG * 4);
    float*  xcur    = (float*) take(SEG * 4);
    float*  ypre    = (float*) take(SEG * 4);
    __bf16* hbuf    = (__bf16*)take((size_t)MROWS * DFF * 2);
    __bf16* wqt     = (__bf16*)take(WSM * 2);
    __bf16* wkt     = (__bf16*)take(WSM * 2);
    __bf16* wvt     = (__bf16*)take(WSM * 2);
    __bf16* wot     = (__bf16*)take(WSM * 2);
    __bf16* l1      = (__bf16*)take(WFF * 2);
    __bf16* l2      = (__bf16*)take(WFF * 2);

    const dim3 blk(256);

    // ---- one-time conversions (bf16 everywhere, convert exactly once) ----
    cvt_bf16_kernel<<<dim3(SEG / 4 / 256), blk, 0, stream>>>(src, src_bf, (int)(SEG / 4));
    whead_kernel<<<dim3(WSM / 256), blk, 0, stream>>>(w_q, wqt);
    whead_kernel<<<dim3(WSM / 256), blk, 0, stream>>>(w_k, wkt);
    whead_kernel<<<dim3(WSM / 256), blk, 0, stream>>>(w_v, wvt);
    wo_t_kernel <<<dim3(WSM / 256), blk, 0, stream>>>(w_o, wot);
    cvt_bf16_kernel<<<dim3(WFF / 4 / 256), blk, 0, stream>>>(lin1_w, l1, (int)(WFF / 4));
    cvt_bf16_kernel<<<dim3(WFF / 4 / 256), blk, 0, stream>>>(lin2_w, l2, (int)(WFF / 4));

    // ---- grids: 64x64 blocks per wave, 8 waves per block ----
    const int gQKV  = (TSEQ / 64) * (DMODEL / 64) / 8;   // 24
    const int gPROJ = (MROWS / 64) * (DMODEL / 64) / 8;  // 96
    const int gFF1  = (MROWS / 64) * (DFF / 64) / 8;     // 384

    // QKV projections: A = src_bf[t, b, :] (lda = B*768), z = batch
    gemm_wmma_kernel<false, false, false, true><<<dim3(gQKV, 1, BBATCH), blk, 0, stream>>>(
        src_bf, BBATCH * DMODEL, DMODEL, wqt, nullptr, nullptr,
        Qws, DMODEL, (long long)TSEQ * DMODEL, TSEQ, DMODEL, DMODEL);
    gemm_wmma_kernel<false, false, false, true><<<dim3(gQKV, 1, BBATCH), blk, 0, stream>>>(
        src_bf, BBATCH * DMODEL, DMODEL, wkt, nullptr, nullptr,
        Kws, DMODEL, (long long)TSEQ * DMODEL, TSEQ, DMODEL, DMODEL);
    gemm_wmma_kernel<false, false, false, true><<<dim3(gQKV, 1, BBATCH), blk, 0, stream>>>(
        src_bf, BBATCH * DMODEL, DMODEL, wvt, nullptr, nullptr,
        Vws, DMODEL, (long long)TSEQ * DMODEL, TSEQ, DMODEL, DMODEL);

    // Banded relative-position attention
    attn_kernel<<<dim3(TSEQ / 16, NHEAD, BBATCH), dim3(32), 0, stream>>>(
        Qws, Kws, Vws, rel_emb, attn);

    // Output projection + residual(src): (4096x768) x (768x768)
    gemm_wmma_kernel<false, false, true, false><<<dim3(gPROJ, 1, 1), blk, 0, stream>>>(
        attn, DMODEL, 0, wot, nullptr, src,
        xpre, DMODEL, 0, MROWS, DMODEL, DMODEL);

    ln_kernel<true><<<dim3(MROWS), blk, 0, stream>>>(xpre, ln1_g, ln1_b, xcur, xbf);

    // FFN1: (4096x768) x lin1^T, +bias, ReLU -> bf16 h
    gemm_wmma_kernel<true, true, false, true><<<dim3(gFF1, 1, 1), blk, 0, stream>>>(
        xbf, DMODEL, 0, l1, lin1_b, nullptr,
        hbuf, DFF, 0, MROWS, DFF, DMODEL);

    // FFN2: (4096x3072) x lin2^T, +bias, +residual(x) -> f32
    gemm_wmma_kernel<false, true, true, false><<<dim3(gPROJ, 1, 1), blk, 0, stream>>>(
        hbuf, DFF, 0, l2, lin2_b, xcur,
        ypre, DMODEL, 0, MROWS, DMODEL, DFF);

    ln_kernel<false><<<dim3(MROWS), blk, 0, stream>>>(ypre, ln2_g, ln2_b, (float*)d_out, nullptr);
}
